// AdaGuidedFilter_17686675325295
// MI455X (gfx1250) — matrix operand order repaired and checked by reading, not compiled, not run on hardware
//
#include <hip/hip_runtime.h>

// Adaptive guided filter, x:(4,64,256,256) f32, r=5, eps=0.01.
// Memory-bound (~134MB min HBM traffic -> ~5.8us @ 23.3TB/s). One pass over HBM.
// Horizontal 11-tap box sums of x and x^2 via V_WMMA_F32_16X16X32_F16 against a
// constant 0/1 banded B matrix; vertical pass + border-normalized epilogue in
// VALU with the strip's x values register-cached from phase 1 (no reread).
// Intermediates in LDS (152KB/WG -> 2 WG/WGP of 320KB).

#define RAD   5
#define EPSF  0.01f
#define WIMG  256
#define HIMG  256
#define STRIP 32                 // output rows per workgroup
#define HALO  (STRIP + 2*RAD)    // 42 live LDS rows
#define LROWS 48                 // padded to 3 row-tiles of 16 (rows 42..47 zero)
#define PH    290                // f16 row pitch (16-col guards both sides; 580B = odd dword stride)
#define PF    260                // f32 row pitch for horizontal-sum buffers

typedef _Float16 v16h __attribute__((ext_vector_type(16)));
typedef float    v8f  __attribute__((ext_vector_type(8)));

__global__ __launch_bounds__(256) void agf_wmma_kernel(const float* __restrict__ x,
                                                       float* __restrict__ out) {
    extern __shared__ char smem[];
    _Float16* xh  = (_Float16*)smem;                 // [LROWS][PH]  f16(x)
    _Float16* xxh = xh + LROWS * PH;                 // [LROWS][PH]  f16(x*x)
    float*    hx  = (float*)(smem + (size_t)2 * LROWS * PH * sizeof(_Float16)); // [LROWS][PF]
    float*    hxx = hx + LROWS * PF;

    const int tid   = threadIdx.x;
    const int plane = blockIdx.x >> 3;       // 8 strips per 256x256 plane
    const int strip = blockIdx.x & 7;
    const int r0    = strip * STRIP;
    const float* __restrict__ xg = x   + (size_t)plane * (HIMG * WIMG);
    float* __restrict__       og = out + (size_t)plane * (HIMG * WIMG);

    // ---- Phase 0: zero the f16 arrays (covers guards + pad rows) ----
    {
        unsigned int* p = (unsigned int*)smem;
        const int nd = (2 * LROWS * PH * 2) / 4;     // 13920 dwords
        for (int i = tid; i < nd; i += 256) p[i] = 0u;
    }
    __syncthreads();

    // ---- Phase 1: coalesced load of strip + halo, convert to f16.
    //      Register-cache the 32 in-strip values this thread's epilogue needs. ----
    float xc[STRIP];
#pragma unroll
    for (int r = 0; r < HALO; ++r) {                 // uniform branch per row
        int gr = r0 - RAD + r;
        float v = (gr >= 0 && gr < HIMG) ? xg[gr * WIMG + tid] : 0.0f;
        if (r >= RAD && r < RAD + STRIP) xc[r - RAD] = v;   // constant index after unroll
        xh [r * PH + 16 + tid] = (_Float16)v;
        xxh[r * PH + 16 + tid] = (_Float16)(v * v);
    }
    __syncthreads();

    // ---- Phase 2: horizontal box sums via WMMA f16 ----
    // Output tile cols n0..n0+15; A covers image cols n0-8..n0+23 (K=32).
    // B[k][n] = 1 iff k in [n+3, n+13]  (i.e. input col in [N-5, N+5]). Constant.
    const int lane = tid & 31;
    const int wv   = tid >> 5;
    const int m    = lane & 15;        // A row / B col / D col-lane
    const int hh   = lane >> 4;        // half-wave
    v16h bm;
#pragma unroll
    for (int v = 0; v < 8; ++v) {
        // B layout: VGPR v<4: K=2v+8h,2v+1+8h ; v>=4: K=16+2(v-4)+8h,17+2(v-4)+8h
        int k0 = (v >> 2) * 16 + hh * 8 + (v & 3) * 2;
        bm[2 * v]     = (_Float16)(((k0     >= m + 3) && (k0     <= m + 13)) ? 1.0f : 0.0f);
        bm[2 * v + 1] = (_Float16)(((k0 + 1 >= m + 3) && (k0 + 1 <= m + 13)) ? 1.0f : 0.0f);
    }

    // 96 tile jobs (3 row-tiles x 16 col-tiles x {x, x^2}), 12 per wave.
    for (int t = 0; t < 12; ++t) {
        int job     = wv * 12 + t;                   // uniform per wave
        int arr     = job / 48;                      // 0 -> x, 1 -> x^2
        int rem     = job % 48;
        int rowtile = rem >> 4;                      // 0..2
        int coltile = rem & 15;                      // 0..15
        // LDS col of K=0: guard(16) + coltile*16 - 8
        const _Float16* src = (arr ? xxh : xh) + (rowtile * 16 + m) * PH + (coltile * 16 + 8);
        v16h a;
#pragma unroll
        for (int i = 0; i < 8; ++i) {
            // A layout: VGPR i<4: K=2i+8h ; i>=4: K=16+2(i-4)+8h
            int g = i >> 2, jj = i & 3;
            int K = g * 16 + hh * 8 + jj * 2;
            a[2 * i]     = src[K];
            a[2 * i + 1] = src[K + 1];
        }
        v8f c = {0.f, 0.f, 0.f, 0.f, 0.f, 0.f, 0.f, 0.f};
        v8f d = __builtin_amdgcn_wmma_f32_16x16x32_f16(false, a, false, bm,
                                                       (short)0, c, false, false);
        float* dst = (arr ? hxx : hx) + coltile * 16 + m;
#pragma unroll
        for (int v = 0; v < 8; ++v)
            dst[(rowtile * 16 + v + 8 * hh) * PF] = d[v];   // D: row v+8h, col lane
    }
    __syncthreads();

    // ---- Phase 3: vertical sliding-window sum + epilogue (thread = column) ----
    const int j = tid;
    float sx = 0.f, sxx = 0.f;
#pragma unroll
    for (int r = 0; r < 2 * RAD; ++r) { sx += hx[r * PF + j]; sxx += hxx[r * PF + j]; }
    const int ncol = min(j + RAD, WIMG - 1) - max(j - RAD, 0) + 1;
#pragma unroll
    for (int i = 0; i < STRIP; ++i) {
        sx  += hx [(i + 2 * RAD) * PF + j];
        sxx += hxx[(i + 2 * RAD) * PF + j];
        int   gr   = r0 + i;
        int   nrow = min(gr + RAD, HIMG - 1) - max(gr - RAD, 0) + 1;
        float invN = 1.0f / (float)(nrow * ncol);
        float mean = sx * invN;
        float var  = sxx * invN - mean * mean;
        float A    = var / (var + EPSF);
        float b    = (1.0f - A) * mean;
        float xv   = xc[i];                          // register-cached, full fp32
        og[gr * WIMG + j] = xv * (A * xv + b);
        sx  -= hx [i * PF + j];
        sxx -= hxx[i * PF + j];
    }
}

extern "C" void kernel_launch(void* const* d_in, const int* in_sizes, int n_in,
                              void* d_out, int out_size, void* d_ws, size_t ws_size,
                              hipStream_t stream) {
    (void)n_in; (void)out_size; (void)d_ws; (void)ws_size;
    const float* x  = (const float*)d_in[0];
    float*       op = (float*)d_out;

    const int planes = in_sizes[0] / (HIMG * WIMG);  // N*C = 256
    const int blocks = planes * (HIMG / STRIP);      // 2048 workgroups

    const size_t shmem = (size_t)2 * LROWS * PH * sizeof(_Float16)
                       + (size_t)2 * LROWS * PF * sizeof(float);   // 155,520 B

    // Allow >64KB dynamic LDS (320KB/WGP on CDNA5); safe to call every time.
    (void)hipFuncSetAttribute((const void*)agf_wmma_kernel,
                              hipFuncAttributeMaxDynamicSharedMemorySize,
                              (int)shmem);

    agf_wmma_kernel<<<blocks, 256, shmem, stream>>>(x, op);
}
